// AttentionHead_41197326303933
// MI455X (gfx1250) — matrix operand order repaired and checked by reading, not compiled, not run on hardware
//
#include <hip/hip_runtime.h>
#include <hip/hip_bf16.h>

typedef __attribute__((ext_vector_type(16))) _Float16 v16h;
typedef __attribute__((ext_vector_type(8)))  float    v8f;

#define EMBED 512
#define HS    32
#define BATCH 4
#define TLEN  4096
#define QW    4            // waves (16-query tiles) per flash-attention workgroup

__device__ __forceinline__ v8f wmma16(v16h a, v16h b, v8f c) {
    // D = A(16x32 f16) * B(32x16 f16) + C(16x16 f32)
    return __builtin_amdgcn_wmma_f32_16x16x32_f16(
        false, a, false, b, (short)0, c, false, false);
}

// ---- CDNA5 async copy global->LDS (ASYNCcnt-tracked) ----
typedef int async_v4i __attribute__((vector_size(16)));
typedef __attribute__((address_space(1))) async_v4i* g_v4i_ptr;
typedef __attribute__((address_space(3))) async_v4i* l_v4i_ptr;

__device__ __forceinline__ void async_load_b128(void* lds_dst, const void* gsrc) {
#if defined(__has_builtin) && __has_builtin(__builtin_amdgcn_global_load_async_to_lds_b128)
    __builtin_amdgcn_global_load_async_to_lds_b128(
        (g_v4i_ptr)gsrc, (l_v4i_ptr)lds_dst, 0, 0);
#else
    unsigned lds_off = (unsigned)(unsigned long long)lds_dst; // low 32 bits = LDS byte addr
    asm volatile("global_load_async_to_lds_b128 %0, %1, off"
                 :: "v"(lds_off), "v"(gsrc) : "memory");
#endif
}
__device__ __forceinline__ void wait_async_le2() { asm volatile("s_wait_asynccnt 0x2" ::: "memory"); }
__device__ __forceinline__ void wait_async_le0() { asm volatile("s_wait_asynccnt 0x0" ::: "memory"); }
// wave-local LDS ordering (no workgroup barrier needed for wave-private scratch)
__device__ __forceinline__ void wave_ds_fence()  { asm volatile("s_wait_dscnt 0x0"    ::: "memory"); }

// ---- Fragment loaders (layouts per CDNA5 ISA 7.12.2) ----

// A fragment: 16x32 f16, row-major source, leading dim ld.
// lane m=lane&15 holds row m; lanes 0-15: K=0..7,16..23; lanes 16-31: K=8..15,24..31.
__device__ __forceinline__ v16h load_a_f16(const _Float16* base, int ld, int lane) {
    int m  = lane & 15;
    int c0 = (lane >> 4) * 8;
    const _Float16* p = base + (size_t)m * ld + c0;
    v16h a;
#pragma unroll
    for (int i = 0; i < 8; ++i) a[i]     = p[i];
#pragma unroll
    for (int i = 0; i < 8; ++i) a[8 + i] = p[16 + i];
    return a;
}

// Same, f32 source converted to f16.
__device__ __forceinline__ v16h load_a_f32cvt(const float* base, int ld, int lane) {
    int m  = lane & 15;
    int c0 = (lane >> 4) * 8;
    const float* p = base + (size_t)m * ld + c0;
    v16h a;
#pragma unroll
    for (int i = 0; i < 8; ++i) a[i]     = (_Float16)p[i];
#pragma unroll
    for (int i = 0; i < 8; ++i) a[8 + i] = (_Float16)p[16 + i];
    return a;
}

// B fragment: 32x16 f16 from column-contiguous storage: cm[n*ldc + k] = B[k][n].
// lane n=lane&15 is the column; lanes 0-15 hold K=0..7,16..23, lanes 16-31 K=8..15,24..31.
__device__ __forceinline__ v16h load_b_colmaj(const _Float16* cm, int ldc, int lane) {
    int n  = lane & 15;
    int k0 = (lane >> 4) * 8;
    const _Float16* p = cm + (size_t)n * ldc + k0;
    v16h b;
#pragma unroll
    for (int i = 0; i < 16; ++i) b[i] = p[(i < 8) ? i : i + 8];
    return b;
}

// ---------------- Kernel 0: transpose/convert all weight matrices to f16 ----------------
// WqT/WkT/WvT: cm[n*512 + k] = W[k*32 + n]   (W is [512][32])
// WpT:         cm[c*32  + r] = Wp[r*512 + c] (Wp is [32][512])
__global__ void prep_weights(const float* __restrict__ Wq, const float* __restrict__ Wk,
                             const float* __restrict__ Wv, const float* __restrict__ Wp,
                             _Float16* __restrict__ WqT, _Float16* __restrict__ WkT,
                             _Float16* __restrict__ WvT, _Float16* __restrict__ WpT) {
    int idx = blockIdx.x * blockDim.x + threadIdx.x;   // 4 * 16384
    int which = idx >> 14;
    int e = idx & 16383;
    if (which < 3) {
        int k = e >> 5, n = e & 31;
        const float* W = (which == 0) ? Wq : (which == 1) ? Wk : Wv;
        _Float16*    D = (which == 0) ? WqT : (which == 1) ? WkT : WvT;
        D[(size_t)n * EMBED + k] = (_Float16)W[e];
    } else {
        int r = e >> 9, c = e & (EMBED - 1);
        WpT[(size_t)c * HS + r] = (_Float16)Wp[e];
    }
}

// ---------------- Kernel 1: q,k,v projections ----------------
// grid.x = B*T/16; blockDim = 192 (6 waves): wave -> {q,k,v} x {cols 0-15,16-31}
__global__ void qkv_proj(const float* __restrict__ x,
                         const _Float16* __restrict__ WqT, const float* __restrict__ bq,
                         const _Float16* __restrict__ WkT, const float* __restrict__ bk,
                         const _Float16* __restrict__ WvT, const float* __restrict__ bv,
                         _Float16* __restrict__ qh, _Float16* __restrict__ kh,
                         _Float16* __restrict__ vT) {
    int wave = threadIdx.x >> 5;
    int lane = threadIdx.x & 31;
    int which = wave >> 1;           // 0=q 1=k 2=v
    int col0  = (wave & 1) * 16;
    const _Float16* Wcm = (which == 0) ? WqT : (which == 1) ? WkT : WvT;
    const float*    bias = (which == 0) ? bq : (which == 1) ? bk : bv;

    size_t row0 = (size_t)blockIdx.x * 16;
    v8f acc = {};
#pragma unroll 4
    for (int c = 0; c < EMBED; c += 32) {
        v16h a = load_a_f32cvt(x + row0 * EMBED + c, EMBED, lane);
        v16h b = load_b_colmaj(Wcm + (size_t)col0 * EMBED + c, EMBED, lane);
        acc = wmma16(a, b, acc);
    }
    int n    = lane & 15;
    int half = (lane >> 4) * 8;
    float bval = bias[col0 + n];
#pragma unroll
    for (int r = 0; r < 8; ++r) {
        size_t gr = row0 + r + half;             // global row in [0, B*T)
        _Float16 hv = (_Float16)(acc[r] + bval);
        if (which == 2) {
            size_t bb = gr / TLEN, t = gr % TLEN;
            vT[(bb * HS + col0 + n) * (size_t)TLEN + t] = hv;  // V transposed
        } else {
            _Float16* dst = (which == 0) ? qh : kh;
            dst[gr * HS + col0 + n] = hv;
        }
    }
}

// ---------------- Kernel 2: fused causal flash attention ----------------
// grid = (T/(16*QW), B), blockDim = 32*QW. A workgroup owns 64 consecutive queries;
// K/V 32-key blocks are async-staged into double-buffered LDS and shared by 4 waves.
__global__ void flash_attn(const _Float16* __restrict__ qh,
                           const _Float16* __restrict__ kh,
                           const _Float16* __restrict__ vT,
                           _Float16* __restrict__ attn_h) {
    __shared__ alignas(16) _Float16 sK[2][32 * 32];   // sK[buf][s*32+d] = k[s0+s][d]
    __shared__ alignas(16) _Float16 sV[2][32 * 32];   // sV[buf][d*32+j] = v[s0+j][d]
    __shared__ alignas(16) float    sS[QW][16 * 32];  // per-wave score scratch
    __shared__ alignas(16) _Float16 sP[QW][16 * 32];  // per-wave P tile (A layout source)
    __shared__ float sAlpha[QW][16];
    __shared__ float sLinv[QW][16];

    int tid  = threadIdx.x;
    int wave = tid >> 5;
    int lane = tid & 31;
    int b    = blockIdx.y;
    int base = blockIdx.x * (16 * QW);
    int t0   = base + wave * 16;
    const float scale = 0.17677669529663687f; // 1/sqrt(32)

    const _Float16* q  = qh + (size_t)b * TLEN * HS;
    const _Float16* k  = kh + (size_t)b * TLEN * HS;
    const _Float16* vt = vT + (size_t)b * HS * TLEN;

    v16h aq = load_a_f16(q + (size_t)t0 * HS, HS, lane);
    v8f acc0 = {}, acc1 = {};
    float m_i = -1e30f, l_i = 0.0f;    // valid in lanes 0..15 (row = lane)

    int n    = lane & 15;
    int half = (lane >> 4) * 8;
    int my_nkb  = (t0 + 47) >> 5;              // blocks this wave needs
    int nkb_max = (base + 16 * QW + 31) >> 5;  // blocks the workgroup stages

    // stage block kb into buffer kb&1: 2 async b128 instructions per wave
    auto stage = [&](int kb) {
        int s0 = kb * 32;
        int buf = kb & 1;
        async_load_b128(&sK[buf][tid * 8], k + (size_t)s0 * HS + tid * 8);
        int d = tid >> 2, part = tid & 3;
        async_load_b128(&sV[buf][d * 32 + part * 8],
                        vt + (size_t)d * TLEN + s0 + part * 8);
    };

    stage(0);
    for (int kb = 0; kb < nkb_max; ++kb) {
        int buf = kb & 1;
        if (kb + 1 < nkb_max) { stage(kb + 1); wait_async_le2(); }
        else                  { wait_async_le0(); }
        __syncthreads();                       // staged K/V visible to all waves

        if (kb < my_nkb) {                     // wave-uniform predicate (EXEC stays full)
            int s0 = kb * 32;
            // scores: two 16x16 WMMA tiles, K/V from LDS
            v16h kb0 = load_b_colmaj(&sK[buf][0],       32, lane);
            v16h kb1 = load_b_colmaj(&sK[buf][16 * 32], 32, lane);
            v8f s0f = {}, s1f = {};
            s0f = wmma16(aq, kb0, s0f);
            s1f = wmma16(aq, kb1, s1f);
#pragma unroll
            for (int r = 0; r < 8; ++r) {
                sS[wave][(r + half) * 32 + n]      = s0f[r];
                sS[wave][(r + half) * 32 + 16 + n] = s1f[r];
            }
            wave_ds_fence();

            // online softmax: lanes 0..15 each own one query row
            if (lane < 16) {
                int t = t0 + lane;
                float* row = &sS[wave][lane * 32];
                int smax = t - s0 + 1;                 // causal: valid keys in block
                if (smax > 32) smax = 32;
                float mblk = -1e30f;
                for (int j = 0; j < smax; ++j) {
                    float v = row[j] * scale;
                    row[j] = v;
                    mblk = fmaxf(mblk, v);
                }
                float mnew  = fmaxf(m_i, mblk);
                float alpha = __expf(m_i - mnew);
                float ssum  = 0.0f;
                _Float16* prow = &sP[wave][lane * 32];
                for (int j = 0; j < 32; ++j) {
                    float p = (j < smax) ? __expf(row[j] - mnew) : 0.0f;
                    ssum += p;
                    prow[j] = (_Float16)p;
                }
                l_i = l_i * alpha + ssum;
                m_i = mnew;
                sAlpha[wave][lane] = alpha;
            }
            wave_ds_fence();

            // rescale running accumulators by per-row alpha
#pragma unroll
            for (int r = 0; r < 8; ++r) {
                float al = sAlpha[wave][r + half];
                acc0[r] *= al;
                acc1[r] *= al;
            }

            // P (16x32 f16 from LDS) x V (32x32 from LDS) -> += attn tile
            v16h pA  = load_a_f16(&sP[wave][0], 32, lane);
            v16h vb0 = load_b_colmaj(&sV[buf][0],       32, lane);
            v16h vb1 = load_b_colmaj(&sV[buf][16 * 32], 32, lane);
            acc0 = wmma16(pA, vb0, acc0);
            acc1 = wmma16(pA, vb1, acc1);
        }
        __syncthreads();   // all waves done with buf before it is restaged
    }

    if (lane < 16) sLinv[wave][lane] = 1.0f / l_i;
    wave_ds_fence();
#pragma unroll
    for (int r = 0; r < 8; ++r) {
        float inv = sLinv[wave][r + half];
        size_t t  = (size_t)t0 + r + half;
        attn_h[((size_t)b * TLEN + t) * HS + n]      = (_Float16)(acc0[r] * inv);
        attn_h[((size_t)b * TLEN + t) * HS + 16 + n] = (_Float16)(acc1[r] * inv);
    }
}

// ---------------- Kernel 3: output projection attn @ Wp + bp ----------------
// grid = (B*T/16, 4), blockDim = 256 (8 waves); wave handles one 16x16 output tile
__global__ void out_proj(const _Float16* __restrict__ attn_h,
                         const _Float16* __restrict__ WpT,
                         const float* __restrict__ bp,
                         float* __restrict__ out) {
    int wave = threadIdx.x >> 5;
    int lane = threadIdx.x & 31;
    int c0   = (blockIdx.y * 8 + wave) * 16;
    size_t row0 = (size_t)blockIdx.x * 16;

    v16h a = load_a_f16(attn_h + row0 * HS, HS, lane);
    v16h b = load_b_colmaj(WpT + (size_t)c0 * HS, HS, lane);
    v8f acc = {};
    acc = wmma16(a, b, acc);

    int n    = lane & 15;
    int half = (lane >> 4) * 8;
    float bval = bp[c0 + n];
#pragma unroll
    for (int r = 0; r < 8; ++r)
        out[(row0 + r + half) * EMBED + c0 + n] = acc[r] + bval;
}

extern "C" void kernel_launch(void* const* d_in, const int* in_sizes, int n_in,
                              void* d_out, int out_size, void* d_ws, size_t ws_size,
                              hipStream_t stream) {
    const float* x  = (const float*)d_in[0];
    const float* Wq = (const float*)d_in[1];
    const float* bq = (const float*)d_in[2];
    const float* Wk = (const float*)d_in[3];
    const float* bk = (const float*)d_in[4];
    const float* Wv = (const float*)d_in[5];
    const float* bv = (const float*)d_in[6];
    const float* Wp = (const float*)d_in[7];
    const float* bp = (const float*)d_in[8];
    float* out = (float*)d_out;

    const size_t NTOK = (size_t)BATCH * TLEN;        // 16384
    _Float16* qh     = (_Float16*)d_ws;              // [B*T][32]
    _Float16* kh     = qh + NTOK * HS;               // [B*T][32]
    _Float16* vT     = kh + NTOK * HS;               // [B][32][T]
    _Float16* attn_h = vT + NTOK * HS;               // [B*T][32]
    _Float16* WqT    = attn_h + NTOK * HS;           // [32 cols][512] col-contig
    _Float16* WkT    = WqT + (size_t)HS * EMBED;
    _Float16* WvT    = WkT + (size_t)HS * EMBED;
    _Float16* WpT    = WvT + (size_t)HS * EMBED;     // [512 cols][32]

    hipLaunchKernelGGL(prep_weights, dim3(4 * HS * EMBED / 256), dim3(256), 0, stream,
                       Wq, Wk, Wv, Wp, WqT, WkT, WvT, WpT);
    hipLaunchKernelGGL(qkv_proj, dim3((unsigned)(NTOK / 16)), dim3(192), 0, stream,
                       x, WqT, bq, WkT, bk, WvT, bv, qh, kh, vT);
    hipLaunchKernelGGL(flash_attn, dim3(TLEN / (16 * QW), BATCH), dim3(32 * QW), 0, stream,
                       qh, kh, vT, attn_h);
    hipLaunchKernelGGL(out_proj, dim3((unsigned)(NTOK / 16), 4), dim3(256), 0, stream,
                       attn_h, WpT, bp, out);
}
